// SelfAttentionLayer_79087527789013
// MI455X (gfx1250) — matrix operand order repaired and checked by reading, compiled, not run on hardware
//
#include <hip/hip_runtime.h>

#define BB 4
#define SS 2048
#define DD 1024

typedef __attribute__((ext_vector_type(16))) __bf16          v16bf;
typedef __attribute__((ext_vector_type(8)))  float           v8f;
typedef __attribute__((ext_vector_type(8)))  unsigned short  us8;
typedef __attribute__((ext_vector_type(16))) unsigned short  us16;
typedef __attribute__((ext_vector_type(4)))  unsigned short  us4;
typedef __attribute__((ext_vector_type(4)))  float           f4;

union US16 { us16 v; us8 h[2]; unsigned short s[16]; };

// fp32 -> bf16, round-to-nearest-even
static __device__ __forceinline__ unsigned short f2bf(float f) {
    unsigned int u = __builtin_bit_cast(unsigned int, f);
    u += 0x7FFFu + ((u >> 16) & 1u);
    return (unsigned short)(u >> 16);
}

// ---- fragment loaders -------------------------------------------------------
// A fragment (16x32 bf16) from row-major bf16 (lda elems/row).
// Lane L: row = L&15, h = L>>4; elems 0..7 -> K = 8h+0..7 ; elems 8..15 -> K = 16+8h+0..7
static __device__ __forceinline__ v16bf load_a_bf16(const unsigned short* A, int lda,
                                                    int row0, int k0, int lane) {
    const int r = lane & 15, h = lane >> 4;
    const unsigned short* p = A + (size_t)(row0 + r) * lda + k0 + 8 * h;
    US16 u;
    u.h[0] = *(const us8*)(p);
    u.h[1] = *(const us8*)(p + 16);
    return __builtin_bit_cast(v16bf, u.v);
}

// B fragment (32x16 bf16) from transposed (N x K row-major) bf16.
// Lane L: col = L&15, h = L>>4; elems 0..7 -> K = 8h+0..7 ; elems 8..15 -> K = 16+8h+0..7
static __device__ __forceinline__ v16bf load_bT(const unsigned short* BT, int ldb,
                                                int col0, int k0, int lane) {
    const int n = lane & 15, h = lane >> 4;
    const unsigned short* p = BT + (size_t)(col0 + n) * ldb + k0 + 8 * h;
    US16 u;
    u.h[0] = *(const us8*)(p);
    u.h[1] = *(const us8*)(p + 16);
    return __builtin_bit_cast(v16bf, u.v);
}

// ---- elementwise convert: x (f32) -> xb (bf16), 4 elems/thread --------------
__global__ void k_convert_x(const float* __restrict__ x, unsigned short* __restrict__ xb) {
    const size_t i = ((size_t)blockIdx.x * blockDim.x + threadIdx.x) * 4;
    const f4 v = *(const f4*)(x + i);
    us4 o;
    o.x = f2bf(v.x); o.y = f2bf(v.y); o.z = f2bf(v.z); o.w = f2bf(v.w);
    *(us4*)(xb + i) = o;
}

// ---- transpose + convert kernels -------------------------------------------
// W (DxD f32 row-major, W[k*D+n]) -> WT (DxD bf16, WT[n*D+k])
__global__ void k_transpose_w(const float* __restrict__ W, unsigned short* __restrict__ WT) {
    __shared__ float tile[32][33];
    const int tx = threadIdx.x, ty = threadIdx.y;
    const int n0 = blockIdx.x * 32, k0 = blockIdx.y * 32;
    tile[ty][tx] = W[(size_t)(k0 + ty) * DD + n0 + tx];
    __syncthreads();
    WT[(size_t)(n0 + ty) * DD + k0 + tx] = f2bf(tile[tx][ty]);
}

// x (B,S,D f32) -> xT (B,D,S bf16)
__global__ void k_transpose_x(const float* __restrict__ x, unsigned short* __restrict__ xT) {
    __shared__ float tile[32][33];
    const int b  = blockIdx.z;
    const int tx = threadIdx.x, ty = threadIdx.y;
    const int d0 = blockIdx.x * 32, s0 = blockIdx.y * 32;
    tile[ty][tx] = x[(size_t)b * SS * DD + (size_t)(s0 + ty) * DD + d0 + tx];
    __syncthreads();
    xT[(size_t)b * DD * SS + (size_t)(d0 + ty) * SS + s0 + tx] = f2bf(tile[tx][ty]);
}

// ---- GEMM 1: Q = x*Wq, K = x*Wk (bf16 in, bf16 out) ------------------------
// one wave -> 32x64 output tile; waves = B*2*(S/32)*(D/64) = 8192
__global__ void k_gemm_qk(const unsigned short* __restrict__ xb,
                          const unsigned short* __restrict__ wqT,
                          const unsigned short* __restrict__ wkT,
                          unsigned short* __restrict__ Qb,
                          unsigned short* __restrict__ Kb) {
    const int lane = threadIdx.x & 31;
    int t = (blockIdx.x * blockDim.x + threadIdx.x) >> 5;
    const int ng = t & 15; t >>= 4;   // D/64 = 16 n-groups
    const int mt = t & 63; t >>= 6;   // S/32 = 64 m-tiles
    const int which = t & 1; t >>= 1; // 0=Q, 1=K
    const int b = t;
    const unsigned short* WT = which ? wkT : wqT;
    const unsigned short* A  = xb + (size_t)b * SS * DD;
    unsigned short* C = (which ? Kb : Qb) + (size_t)b * SS * DD;
    const int m0 = mt * 32, n0 = ng * 64;

    const v8f zero = {0.f,0.f,0.f,0.f,0.f,0.f,0.f,0.f};
    v8f acc0[4], acc1[4];
#pragma unroll
    for (int i = 0; i < 4; ++i) { acc0[i] = zero; acc1[i] = zero; }

    for (int k0 = 0; k0 < DD; k0 += 32) {
        const v16bf a0 = load_a_bf16(A, DD, m0,      k0, lane);
        const v16bf a1 = load_a_bf16(A, DD, m0 + 16, k0, lane);
#pragma unroll
        for (int ti = 0; ti < 4; ++ti) {
            const v16bf bf = load_bT(WT, DD, n0 + 16 * ti, k0, lane);
            acc0[ti] = __builtin_amdgcn_wmma_f32_16x16x32_bf16(
                false, a0, false, bf, (short)0, acc0[ti], false, false);
            acc1[ti] = __builtin_amdgcn_wmma_f32_16x16x32_bf16(
                false, a1, false, bf, (short)0, acc1[ti], false, false);
        }
    }
    const int n = lane & 15, h = lane >> 4;
#pragma unroll
    for (int ti = 0; ti < 4; ++ti)
#pragma unroll
        for (int r = 0; r < 8; ++r) {
            C[(size_t)(m0      + r + 8 * h) * DD + n0 + 16 * ti + n] = f2bf(acc0[ti][r]);
            C[(size_t)(m0 + 16 + r + 8 * h) * DD + n0 + 16 * ti + n] = f2bf(acc1[ti][r]);
        }
}

// ---- GEMM 2: scores = Q * K^T (f32 out) ------------------------------------
// one wave -> 32x64; waves = B*(S/32)*(S/64) = 8192
__global__ void k_gemm_scores(const unsigned short* __restrict__ Qb,
                              const unsigned short* __restrict__ Kb,
                              float* __restrict__ scores) {
    const int lane = threadIdx.x & 31;
    int t = (blockIdx.x * blockDim.x + threadIdx.x) >> 5;
    const int ng = t & 31; t >>= 5;   // S/64 = 32
    const int mt = t & 63; t >>= 6;   // S/32 = 64
    const int b = t;
    const unsigned short* Q = Qb + (size_t)b * SS * DD;
    const unsigned short* K = Kb + (size_t)b * SS * DD; // row-major SxD == (NxK) for B-frag
    float* Sc = scores + (size_t)b * SS * SS;
    const int m0 = mt * 32, n0 = ng * 64;

    const v8f zero = {0.f,0.f,0.f,0.f,0.f,0.f,0.f,0.f};
    v8f acc0[4], acc1[4];
#pragma unroll
    for (int i = 0; i < 4; ++i) { acc0[i] = zero; acc1[i] = zero; }

    for (int k0 = 0; k0 < DD; k0 += 32) {
        const v16bf a0 = load_a_bf16(Q, DD, m0,      k0, lane);
        const v16bf a1 = load_a_bf16(Q, DD, m0 + 16, k0, lane);
#pragma unroll
        for (int ti = 0; ti < 4; ++ti) {
            const v16bf bf = load_bT(K, DD, n0 + 16 * ti, k0, lane);
            acc0[ti] = __builtin_amdgcn_wmma_f32_16x16x32_bf16(
                false, a0, false, bf, (short)0, acc0[ti], false, false);
            acc1[ti] = __builtin_amdgcn_wmma_f32_16x16x32_bf16(
                false, a1, false, bf, (short)0, acc1[ti], false, false);
        }
    }
    const int n = lane & 15, h = lane >> 4;
#pragma unroll
    for (int ti = 0; ti < 4; ++ti)
#pragma unroll
        for (int r = 0; r < 8; ++r) {
            Sc[(size_t)(m0      + r + 8 * h) * SS + n0 + 16 * ti + n] = acc0[ti][r];
            Sc[(size_t)(m0 + 16 + r + 8 * h) * SS + n0 + 16 * ti + n] = acc1[ti][r];
        }
}

// ---- softmax: one wave per row; writes normalized bf16 probabilities -------
__global__ void k_softmax_p(const float* __restrict__ scores,
                            unsigned short* __restrict__ Pb) {
    const int lane = threadIdx.x & 31;
    const int row = (blockIdx.x * blockDim.x + threadIdx.x) >> 5; // 0 .. B*S-1
    const float* p = scores + (size_t)row * SS;
    unsigned short* q = Pb + (size_t)row * SS;

    float m = -3.402823466e38f;
    for (int i = lane * 4; i < SS; i += 128) {
        const f4 v = *(const f4*)(p + i);
        m = fmaxf(m, fmaxf(fmaxf(v.x, v.y), fmaxf(v.z, v.w)));
    }
#pragma unroll
    for (int off = 16; off; off >>= 1) m = fmaxf(m, __shfl_xor(m, off, 32));

    float s = 0.f;
    for (int i = lane * 4; i < SS; i += 128) {
        const f4 v = *(const f4*)(p + i);
        s += __expf(v.x - m) + __expf(v.y - m) + __expf(v.z - m) + __expf(v.w - m);
    }
#pragma unroll
    for (int off = 16; off; off >>= 1) s += __shfl_xor(s, off, 32);
    const float ri = 1.f / s;

    for (int i = lane * 4; i < SS; i += 128) {
        const f4 v = *(const f4*)(p + i);
        us4 o;
        o.x = f2bf(__expf(v.x - m) * ri);
        o.y = f2bf(__expf(v.y - m) * ri);
        o.z = f2bf(__expf(v.z - m) * ri);
        o.w = f2bf(__expf(v.w - m) * ri);
        *(us4*)(q + i) = o;
    }
}

// ---- GEMM 3: out = P * x (bf16 in, f32 out) --------------------------------
// one wave -> 32x64; waves = B*(S/32)*(D/64) = 4096
__global__ void k_gemm_out(const unsigned short* __restrict__ Pb,
                           const unsigned short* __restrict__ xT,
                           float* __restrict__ out) {
    const int lane = threadIdx.x & 31;
    int t = (blockIdx.x * blockDim.x + threadIdx.x) >> 5;
    const int ng = t & 15; t >>= 4;   // D/64 = 16
    const int mt = t & 63; t >>= 6;   // S/32 = 64
    const int b = t;
    const unsigned short* P = Pb + (size_t)b * SS * SS;
    const unsigned short* V = xT + (size_t)b * DD * SS;  // V[n*SS + k]
    float* O = out + (size_t)b * SS * DD;
    const int m0 = mt * 32, n0 = ng * 64;

    const v8f zero = {0.f,0.f,0.f,0.f,0.f,0.f,0.f,0.f};
    v8f acc0[4], acc1[4];
#pragma unroll
    for (int i = 0; i < 4; ++i) { acc0[i] = zero; acc1[i] = zero; }

    for (int k0 = 0; k0 < SS; k0 += 32) {
        const v16bf a0 = load_a_bf16(P, SS, m0,      k0, lane);
        const v16bf a1 = load_a_bf16(P, SS, m0 + 16, k0, lane);
#pragma unroll
        for (int ti = 0; ti < 4; ++ti) {
            const v16bf bf = load_bT(V, SS, n0 + 16 * ti, k0, lane);
            acc0[ti] = __builtin_amdgcn_wmma_f32_16x16x32_bf16(
                false, a0, false, bf, (short)0, acc0[ti], false, false);
            acc1[ti] = __builtin_amdgcn_wmma_f32_16x16x32_bf16(
                false, a1, false, bf, (short)0, acc1[ti], false, false);
        }
    }
    const int n = lane & 15, h = lane >> 4;
#pragma unroll
    for (int ti = 0; ti < 4; ++ti)
#pragma unroll
        for (int r = 0; r < 8; ++r) {
            O[(size_t)(m0      + r + 8 * h) * DD + n0 + 16 * ti + n] = acc0[ti][r];
            O[(size_t)(m0 + 16 + r + 8 * h) * DD + n0 + 16 * ti + n] = acc1[ti][r];
        }
}

// ---- launcher ---------------------------------------------------------------
extern "C" void kernel_launch(void* const* d_in, const int* in_sizes, int n_in,
                              void* d_out, int out_size, void* d_ws, size_t ws_size,
                              hipStream_t stream) {
    const float* x  = (const float*)d_in[0];
    const float* Wq = (const float*)d_in[1];
    const float* Wk = (const float*)d_in[2];
    float* out = (float*)d_out;

    // workspace carve-up (~164 MB total)
    char* p = (char*)d_ws;
    unsigned short* wqT = (unsigned short*)p; p += (size_t)DD * DD * 2;       //  2 MB
    unsigned short* wkT = (unsigned short*)p; p += (size_t)DD * DD * 2;       //  2 MB
    unsigned short* xb  = (unsigned short*)p; p += (size_t)BB * SS * DD * 2;  // 16 MB
    unsigned short* Qb  = (unsigned short*)p; p += (size_t)BB * SS * DD * 2;  // 16 MB
    unsigned short* Kb  = (unsigned short*)p; p += (size_t)BB * SS * DD * 2;  // 16 MB
    unsigned short* xT  = (unsigned short*)p; p += (size_t)BB * DD * SS * 2;  // 16 MB
    float*          Sc  = (float*)p;          p += (size_t)BB * SS * SS * 4;  // 64 MB
    unsigned short* Pb  = (unsigned short*)p;                                 // 32 MB

    const dim3 blk32(32, 32);
    k_convert_x<<<(BB * SS * DD / 4) / 256, 256, 0, stream>>>(x, xb);
    k_transpose_w<<<dim3(DD / 32, DD / 32), blk32, 0, stream>>>(Wq, wqT);
    k_transpose_w<<<dim3(DD / 32, DD / 32), blk32, 0, stream>>>(Wk, wkT);
    k_transpose_x<<<dim3(DD / 32, SS / 32, BB), blk32, 0, stream>>>(x, xT);

    k_gemm_qk<<<1024, 256, 0, stream>>>(xb, wqT, wkT, Qb, Kb);   // 8192 waves
    k_gemm_scores<<<1024, 256, 0, stream>>>(Qb, Kb, Sc);         // 8192 waves
    k_softmax_p<<<(BB * SS) / 8, 256, 0, stream>>>(Sc, Pb);
    k_gemm_out<<<512, 256, 0, stream>>>(Pb, xT, out);            // 4096 waves
}
